// AttentionConvNet_63402307224308
// MI455X (gfx1250) — matrix operand order repaired and checked by reading, not compiled, not run on hardware
//
#include <hip/hip_runtime.h>
#include <math.h>

typedef __attribute__((ext_vector_type(16))) _Float16 v16h;
typedef __attribute__((ext_vector_type(8)))  float    v8f;
typedef _Float16 h16;

#define NNODES 100000
#define NEDGES 1600000
#define NGRAPH 4096
#define FIN    92
#define HID    51
#define EDIM   10
#define TSTEPS 7
#define FINP   96     // FIN padded to multiple of 32
#define HP     64     // HID / (HID+EDIM) padded K
#define G3     153    // 3*HID
#define G3P    160    // padded N for GRU gate GEMMs

// ---------------- math helpers ----------------
__device__ __forceinline__ float lrelu(float v){ return v > 0.f ? v : 0.01f * v; }
__device__ __forceinline__ float eluf (float v){ return v > 0.f ? v : (__expf(v) - 1.f); }
__device__ __forceinline__ float sigm (float v){ return 1.f / (1.f + __expf(-v)); }

// monotone float<->uint map for atomicMax-based segment max
__device__ __forceinline__ unsigned f2ord(float f){
  unsigned u = __float_as_uint(f);
  return (u & 0x80000000u) ? ~u : (u | 0x80000000u);
}
__device__ __forceinline__ float ord2f(unsigned u){
  return (u & 0x80000000u) ? __uint_as_float(u & 0x7fffffffu) : __uint_as_float(~u);
}

__device__ __forceinline__ v8f vzero8(){
  v8f z;
#pragma unroll
  for (int i = 0; i < 8; ++i) z[i] = 0.f;
  return z;
}

// ---------------- WMMA fragment loaders (wave32, 16x16x32 f16) ----------------
// A fragment: row-major f16, A points at (row0, k=kc), leading dim ldk.
// Lane l: m = l&15; koff = (l>=16)?16:0; halves j=0..15 -> K=koff+j.
__device__ __forceinline__ v16h load_a_frag(const h16* A, int ldk){
  int lane = threadIdx.x & 31;
  int m    = lane & 15;
  int koff = (lane >> 4) ? 16 : 0;
  const h16* p = A + m * ldk + koff;
  v16h r;
#pragma unroll
  for (int i = 0; i < 16; ++i) r[i] = p[i];
  return r;
}

// B fragment: weights pre-packed as Bp[n][k] (n-major, Kpad contiguous halves per column).
// Caller pre-offsets Bp by (tile_n*16)*Kpad + kc. Lane l: n = l&15; koff = (l>=16)?16:0.
__device__ __forceinline__ v16h load_b_frag(const h16* Bp, int Kpad){
  int lane = threadIdx.x & 31;
  int n    = lane & 15;
  int koff = (lane >> 4) ? 16 : 0;
  const h16* p = Bp + n * Kpad + koff;
  v16h r;
#pragma unroll
  for (int i = 0; i < 16; ++i) r[i] = p[i];
  return r;
}

__device__ __forceinline__ v8f wmma16(v16h a, v16h b, v8f c){
  return __builtin_amdgcn_wmma_f32_16x16x32_f16(false, a, false, b, (short)0, c, false, false);
}

// ---------------- weight pre-pack: W[K][Nout] f32 -> Bp[Npad][Kpad] f16 ----------------
__global__ void pack_weight_kernel(const float* __restrict__ W, h16* __restrict__ Bp,
                                   int K, int Nout, int Kpad, int Npad){
  int idx = blockIdx.x * blockDim.x + threadIdx.x;
  if (idx >= Kpad * Npad) return;
  int n = idx / Kpad, k = idx % Kpad;
  float v = (k < K && n < Nout) ? W[k * Nout + n] : 0.f;
  Bp[idx] = (h16)v;
}

// ---------------- init: h = gate_bias, segment max/sum, outsum ----------------
__global__ void init_node_kernel(float* __restrict__ hN, const float* __restrict__ gate_bias,
                                 unsigned* __restrict__ amaxN, float* __restrict__ esumN,
                                 float* __restrict__ outsum){
  int idx = blockIdx.x * blockDim.x + threadIdx.x;
  if (idx < NNODES * HID) hN[idx] = gate_bias[idx % HID];
  if (idx < NNODES) { amaxN[idx] = f2ord(-__builtin_inff()); esumN[idx] = 0.f; }
  if (idx < NGRAPH * HID) outsum[idx] = 0.f;
}

// ---------------- lin1: x = lrelu(x @ W1 + b1) ----------------
__global__ void __launch_bounds__(32) lin1_kernel(const float* __restrict__ x,
                                                  const h16* __restrict__ W1p,
                                                  const float* __restrict__ b1,
                                                  float* __restrict__ xh,
                                                  h16* __restrict__ xh16){
  __shared__ h16 As[16 * FINP];
  int lane = threadIdx.x;
  int row0 = blockIdx.x * 16;
  for (int idx = lane; idx < 16 * FINP; idx += 32) {
    int r = idx / FINP, c = idx % FINP;
    int node = row0 + r;
    float v = (node < NNODES && c < FIN) ? x[(size_t)node * FIN + c] : 0.f;
    As[idx] = (h16)v;
  }
  __syncthreads();
  v8f acc[4]; 
#pragma unroll
  for (int t = 0; t < 4; ++t) acc[t] = vzero8();
  for (int kc = 0; kc < FINP; kc += 32) {
    v16h a = load_a_frag(As + kc, FINP);
#pragma unroll
    for (int t = 0; t < 4; ++t) {
      v16h b = load_b_frag(W1p + (t * 16) * FINP + kc, FINP);
      acc[t] = wmma16(a, b, acc[t]);
    }
  }
  int nlo = lane & 15, hi = lane >> 4;
#pragma unroll
  for (int t = 0; t < 4; ++t)
#pragma unroll
    for (int v = 0; v < 8; ++v) {
      int m = v + hi * 8, n = t * 16 + nlo;
      int node = row0 + m;
      if (node < NNODES) {
        float val = acc[t][v] + (n < HID ? b1[n] : 0.f);
        val = lrelu(val);
        if (n < HID) xh[(size_t)node * HID + n] = val;
        xh16[(size_t)node * HP + n] = (h16)(n < HID ? val : 0.f);
      }
    }
}

// ---------------- edge pass 1: a_e = lrelu(xe.att_l + x[dst].att_r); segment max ----------------
__global__ void __launch_bounds__(32) edge_a_kernel(const h16* __restrict__ xh16,
                                                    const float* __restrict__ xh,
                                                    const float* __restrict__ edge_attr,
                                                    const int* __restrict__ ei,
                                                    const h16* __restrict__ We1p,
                                                    const float* __restrict__ att_l,
                                                    const float* __restrict__ att_r,
                                                    float* __restrict__ ae,
                                                    unsigned* __restrict__ amaxN){
  __shared__ h16 As[16 * HP];
  __shared__ int sE[16], dE[16];
  __shared__ float rowsum[16];
  int lane = threadIdx.x;
  int e0 = blockIdx.x * 16;
  const int* src = ei;
  const int* dst = ei + NEDGES;
  if (lane < 16) {
    int e = e0 + lane;
    sE[lane] = (e < NEDGES) ? src[e] : 0;
    dE[lane] = (e < NEDGES) ? dst[e] : 0;
  }
  __syncthreads();
  for (int idx = lane; idx < 16 * HP; idx += 32) {
    int r = idx / HP, c = idx % HP;
    int e = e0 + r;
    h16 v = (h16)0.f;
    if (e < NEDGES) {
      if (c < HID)            v = xh16[(size_t)sE[r] * HP + c];
      else if (c < HID + EDIM) v = (h16)edge_attr[(size_t)e * EDIM + (c - HID)];
    }
    As[idx] = v;
  }
  __syncthreads();
  v8f acc[4];
#pragma unroll
  for (int t = 0; t < 4; ++t) acc[t] = vzero8();
  for (int kc = 0; kc < HP; kc += 32) {
    v16h a = load_a_frag(As + kc, HP);
#pragma unroll
    for (int t = 0; t < 4; ++t) {
      v16h b = load_b_frag(We1p + (t * 16) * HP + kc, HP);
      acc[t] = wmma16(a, b, acc[t]);
    }
  }
  int nlo = lane & 15, hi = lane >> 4;
  float p[8];
#pragma unroll
  for (int v = 0; v < 8; ++v) p[v] = 0.f;
#pragma unroll
  for (int t = 0; t < 4; ++t) {
    int n = t * 16 + nlo;
    float aln = (n < HID) ? att_l[n] : 0.f;
#pragma unroll
    for (int v = 0; v < 8; ++v) p[v] += lrelu(acc[t][v]) * aln;
  }
  for (int off = 8; off > 0; off >>= 1)
#pragma unroll
    for (int v = 0; v < 8; ++v) p[v] += __shfl_xor(p[v], off, 32);
  if ((lane & 15) == 0)
#pragma unroll
    for (int v = 0; v < 8; ++v) rowsum[hi * 8 + v] = p[v];
  __syncthreads();
  if (lane < 16) {
    int e = e0 + lane;
    if (e < NEDGES) {
      int d = dE[lane];
      float s = rowsum[lane];
      for (int c = 0; c < HID; ++c) s += xh[(size_t)d * HID + c] * att_r[c];
      s = lrelu(s);
      ae[e] = s;
      atomicMax(&amaxN[d], f2ord(s));
    }
  }
}

// ---------------- edge pass 1b: exp + segment sum ----------------
__global__ void edge_exp_kernel(float* __restrict__ ae, const int* __restrict__ ei,
                                const unsigned* __restrict__ amaxN, float* __restrict__ esumN){
  int e = blockIdx.x * blockDim.x + threadIdx.x;
  if (e >= NEDGES) return;
  int d = ei[NEDGES + e];
  float ex = __expf(ae[e] - ord2f(amaxN[d]));
  ae[e] = ex;
  atomicAdd(&esumN[d], ex);
}

// ---------------- edge pass 2: recompute xe, m = xe@We2, scatter m*alpha to h[dst] ----------------
__global__ void __launch_bounds__(32) edge_scatter_kernel(const h16* __restrict__ xh16,
                                                          const float* __restrict__ edge_attr,
                                                          const int* __restrict__ ei,
                                                          const h16* __restrict__ We1p,
                                                          const h16* __restrict__ We2p,
                                                          const float* __restrict__ ae,
                                                          const float* __restrict__ esumN,
                                                          float* __restrict__ hN){
  __shared__ h16 As[16 * HP];
  __shared__ h16 Xe[16 * HP];
  __shared__ int sE[16], dE[16];
  __shared__ float alph[16];
  int lane = threadIdx.x;
  int e0 = blockIdx.x * 16;
  const int* src = ei;
  const int* dst = ei + NEDGES;
  if (lane < 16) {
    int e = e0 + lane;
    sE[lane] = (e < NEDGES) ? src[e] : 0;
    dE[lane] = (e < NEDGES) ? dst[e] : 0;
  }
  __syncthreads();
  for (int idx = lane; idx < 16 * HP; idx += 32) {
    int r = idx / HP, c = idx % HP;
    int e = e0 + r;
    h16 v = (h16)0.f;
    if (e < NEDGES) {
      if (c < HID)            v = xh16[(size_t)sE[r] * HP + c];
      else if (c < HID + EDIM) v = (h16)edge_attr[(size_t)e * EDIM + (c - HID)];
    }
    As[idx] = v;
  }
  __syncthreads();
  v8f acc[4];
#pragma unroll
  for (int t = 0; t < 4; ++t) acc[t] = vzero8();
  for (int kc = 0; kc < HP; kc += 32) {
    v16h a = load_a_frag(As + kc, HP);
#pragma unroll
    for (int t = 0; t < 4; ++t) {
      v16h b = load_b_frag(We1p + (t * 16) * HP + kc, HP);
      acc[t] = wmma16(a, b, acc[t]);
    }
  }
  int nlo = lane & 15, hi = lane >> 4;
#pragma unroll
  for (int t = 0; t < 4; ++t)
#pragma unroll
    for (int v = 0; v < 8; ++v) {
      int m = v + hi * 8, n = t * 16 + nlo;
      Xe[m * HP + n] = (h16)lrelu(acc[t][v]);
    }
  if (lane < 16) {
    int e = e0 + lane;
    alph[lane] = (e < NEDGES) ? ae[e] / (esumN[dE[lane]] + 1e-16f) : 0.f;
  }
  __syncthreads();
  v8f acc2[4];
#pragma unroll
  for (int t = 0; t < 4; ++t) acc2[t] = vzero8();
  for (int kc = 0; kc < HP; kc += 32) {
    v16h a = load_a_frag(Xe + kc, HP);
#pragma unroll
    for (int t = 0; t < 4; ++t) {
      v16h b = load_b_frag(We2p + (t * 16) * HP + kc, HP);
      acc2[t] = wmma16(a, b, acc2[t]);
    }
  }
#pragma unroll
  for (int t = 0; t < 4; ++t)
#pragma unroll
    for (int v = 0; v < 8; ++v) {
      int m = v + hi * 8, n = t * 16 + nlo;
      int e = e0 + m;
      if (e < NEDGES && n < HID)
        atomicAdd(&hN[(size_t)dE[m] * HID + n], acc2[t][v] * alph[m]);
    }
}

// ---------------- GRU1: x = relu(GRU(elu(h), x)); also outsum[batch] += x ----------------
__global__ void __launch_bounds__(32) gru1_kernel(const float* __restrict__ hN,
                                                  float* __restrict__ xh,
                                                  h16* __restrict__ xh16,
                                                  const h16* __restrict__ Wih1p,
                                                  const float* __restrict__ bih1,
                                                  const h16* __restrict__ Whh1p,
                                                  const float* __restrict__ bhh1,
                                                  const int* __restrict__ batch,
                                                  float* __restrict__ outsum){
  __shared__ h16 Ah[16 * HP];
  __shared__ float Gi[16 * G3P];
  __shared__ float Gh[16 * G3P];
  int lane = threadIdx.x;
  int row0 = blockIdx.x * 16;
  for (int idx = lane; idx < 16 * HP; idx += 32) {
    int r = idx / HP, c = idx % HP;
    int node = row0 + r;
    float v = (node < NNODES && c < HID) ? eluf(hN[(size_t)node * HID + c]) : 0.f;
    Ah[idx] = (h16)v;
  }
  __syncthreads();
  const h16* Ax = xh16 + (size_t)row0 * HP;
  int nlo = lane & 15, hi = lane >> 4;
  for (int t = 0; t < 10; ++t) {
    v8f a1 = vzero8(), a2 = vzero8();
    for (int kc = 0; kc < HP; kc += 32) {
      v16h af = load_a_frag(Ah + kc, HP);
      v16h bf = load_b_frag(Wih1p + (t * 16) * HP + kc, HP);
      a1 = wmma16(af, bf, a1);
      v16h af2 = load_a_frag(Ax + kc, HP);
      v16h bf2 = load_b_frag(Whh1p + (t * 16) * HP + kc, HP);
      a2 = wmma16(af2, bf2, a2);
    }
#pragma unroll
    for (int v = 0; v < 8; ++v) {
      int m = v + hi * 8, n = t * 16 + nlo;
      Gi[m * G3P + n] = a1[v] + (n < G3 ? bih1[n] : 0.f);
      Gh[m * G3P + n] = a2[v] + (n < G3 ? bhh1[n] : 0.f);
    }
  }
  __syncthreads();
  for (int idx = lane; idx < 16 * HID; idx += 32) {
    int m = idx / HID, c = idx % HID;
    int node = row0 + m;
    if (node < NNODES) {
      float ir = Gi[m * G3P + c], iz = Gi[m * G3P + HID + c], in = Gi[m * G3P + 2 * HID + c];
      float hr = Gh[m * G3P + c], hz = Gh[m * G3P + HID + c], hn = Gh[m * G3P + 2 * HID + c];
      float r = sigm(ir + hr), z = sigm(iz + hz);
      float nn = tanhf(in + r * hn);
      float hp = xh[(size_t)node * HID + c];
      float v = fmaxf((1.f - z) * nn + z * hp, 0.f);
      xh[(size_t)node * HID + c]  = v;
      xh16[(size_t)node * HP + c] = (h16)v;
      atomicAdd(&outsum[(size_t)batch[node] * HID + c], v);
    }
  }
}

__global__ void relu_out_kernel(const float* __restrict__ outsum, float* __restrict__ outG){
  int idx = blockIdx.x * blockDim.x + threadIdx.x;
  if (idx < NGRAPH * HID) outG[idx] = fmaxf(outsum[idx], 0.f);
}

// ---------------- xl = x @ Wm ; al = xl . att_src ----------------
__global__ void __launch_bounds__(32) xl_kernel(const h16* __restrict__ xh16,
                                                const h16* __restrict__ Wmp,
                                                const float* __restrict__ att_src,
                                                float* __restrict__ xl,
                                                float* __restrict__ al){
  __shared__ float rowsum[16];
  int lane = threadIdx.x;
  int row0 = blockIdx.x * 16;
  const h16* Ax = xh16 + (size_t)row0 * HP;
  v8f acc[4];
#pragma unroll
  for (int t = 0; t < 4; ++t) acc[t] = vzero8();
  for (int kc = 0; kc < HP; kc += 32) {
    v16h a = load_a_frag(Ax + kc, HP);
#pragma unroll
    for (int t = 0; t < 4; ++t) {
      v16h b = load_b_frag(Wmp + (t * 16) * HP + kc, HP);
      acc[t] = wmma16(a, b, acc[t]);
    }
  }
  int nlo = lane & 15, hi = lane >> 4;
  float p[8];
#pragma unroll
  for (int v = 0; v < 8; ++v) p[v] = 0.f;
#pragma unroll
  for (int t = 0; t < 4; ++t) {
    int n = t * 16 + nlo;
    float as = (n < HID) ? att_src[n] : 0.f;
#pragma unroll
    for (int v = 0; v < 8; ++v) {
      int m = v + hi * 8;
      int node = row0 + m;
      if (node < NNODES && n < HID) xl[(size_t)node * HID + n] = acc[t][v];
      p[v] += acc[t][v] * as;
    }
  }
  for (int off = 8; off > 0; off >>= 1)
#pragma unroll
    for (int v = 0; v < 8; ++v) p[v] += __shfl_xor(p[v], off, 32);
  if ((lane & 15) == 0)
#pragma unroll
    for (int v = 0; v < 8; ++v) rowsum[hi * 8 + v] = p[v];
  __syncthreads();
  if (lane < 16) {
    int node = row0 + lane;
    if (node < NNODES) al[node] = rowsum[lane];
  }
}

// ---------------- per-timestep kernels ----------------
__global__ void tinit_kernel(unsigned* __restrict__ amaxB, float* __restrict__ esumB,
                             float* __restrict__ hmB, const float* __restrict__ mol_bias){
  int idx = blockIdx.x * blockDim.x + threadIdx.x;
  if (idx < NGRAPH * HID) hmB[idx] = mol_bias[idx % HID];
  if (idx < NGRAPH) { amaxB[idx] = f2ord(-__builtin_inff()); esumB[idx] = 0.f; }
}

__global__ void __launch_bounds__(32) ar_kernel(const float* __restrict__ outG,
                                                const h16* __restrict__ Wmp,
                                                const float* __restrict__ att_dst,
                                                float* __restrict__ arG){
  __shared__ h16 Ao[16 * HP];
  __shared__ float rowsum[16];
  int lane = threadIdx.x;
  int row0 = blockIdx.x * 16;
  for (int idx = lane; idx < 16 * HP; idx += 32) {
    int r = idx / HP, c = idx % HP;
    float v = (c < HID) ? outG[(size_t)(row0 + r) * HID + c] : 0.f;
    Ao[idx] = (h16)v;
  }
  __syncthreads();
  v8f acc[4];
#pragma unroll
  for (int t = 0; t < 4; ++t) acc[t] = vzero8();
  for (int kc = 0; kc < HP; kc += 32) {
    v16h a = load_a_frag(Ao + kc, HP);
#pragma unroll
    for (int t = 0; t < 4; ++t) {
      v16h b = load_b_frag(Wmp + (t * 16) * HP + kc, HP);
      acc[t] = wmma16(a, b, acc[t]);
    }
  }
  int nlo = lane & 15, hi = lane >> 4;
  float p[8];
#pragma unroll
  for (int v = 0; v < 8; ++v) p[v] = 0.f;
#pragma unroll
  for (int t = 0; t < 4; ++t) {
    int n = t * 16 + nlo;
    float ad = (n < HID) ? att_dst[n] : 0.f;
#pragma unroll
    for (int v = 0; v < 8; ++v) p[v] += acc[t][v] * ad;
  }
  for (int off = 8; off > 0; off >>= 1)
#pragma unroll
    for (int v = 0; v < 8; ++v) p[v] += __shfl_xor(p[v], off, 32);
  if ((lane & 15) == 0)
#pragma unroll
    for (int v = 0; v < 8; ++v) rowsum[hi * 8 + v] = p[v];
  __syncthreads();
  if (lane < 16) arG[row0 + lane] = rowsum[lane];
}

__global__ void node_a_kernel(const float* __restrict__ al, const float* __restrict__ arG,
                              const int* __restrict__ batch, float* __restrict__ an,
                              unsigned* __restrict__ amaxB){
  int n = blockIdx.x * blockDim.x + threadIdx.x;
  if (n >= NNODES) return;
  int b = batch[n];
  float a = lrelu(al[n] + arG[b]);
  an[n] = a;
  atomicMax(&amaxB[b], f2ord(a));
}

__global__ void node_exp_kernel(float* __restrict__ an, const int* __restrict__ batch,
                                const unsigned* __restrict__ amaxB, float* __restrict__ esumB){
  int n = blockIdx.x * blockDim.x + threadIdx.x;
  if (n >= NNODES) return;
  int b = batch[n];
  float ex = __expf(an[n] - ord2f(amaxB[b]));
  an[n] = ex;
  atomicAdd(&esumB[b], ex);
}

__global__ void hm_kernel(const float* __restrict__ an, const int* __restrict__ batch,
                          const float* __restrict__ esumB, const float* __restrict__ xl,
                          float* __restrict__ hmB){
  int idx = blockIdx.x * blockDim.x + threadIdx.x;
  if (idx >= NNODES * HID) return;
  int n = idx / HID, c = idx % HID;
  int b = batch[n];
  float alpha = an[n] / (esumB[b] + 1e-16f);
  atomicAdd(&hmB[(size_t)b * HID + c], xl[idx] * alpha);
}

__global__ void __launch_bounds__(32) gru2_kernel(const float* __restrict__ hmB,
                                                  float* __restrict__ outG,
                                                  const h16* __restrict__ Wih2p,
                                                  const float* __restrict__ bih2,
                                                  const h16* __restrict__ Whh2p,
                                                  const float* __restrict__ bhh2){
  __shared__ h16 Ah[16 * HP];
  __shared__ h16 Ao[16 * HP];
  __shared__ float Gi[16 * G3P];
  __shared__ float Gh[16 * G3P];
  int lane = threadIdx.x;
  int row0 = blockIdx.x * 16;
  for (int idx = lane; idx < 16 * HP; idx += 32) {
    int r = idx / HP, c = idx % HP;
    size_t g = (size_t)(row0 + r);
    Ah[idx] = (h16)((c < HID) ? eluf(hmB[g * HID + c]) : 0.f);
    Ao[idx] = (h16)((c < HID) ? outG[g * HID + c] : 0.f);
  }
  __syncthreads();
  int nlo = lane & 15, hi = lane >> 4;
  for (int t = 0; t < 10; ++t) {
    v8f a1 = vzero8(), a2 = vzero8();
    for (int kc = 0; kc < HP; kc += 32) {
      v16h af = load_a_frag(Ah + kc, HP);
      v16h bf = load_b_frag(Wih2p + (t * 16) * HP + kc, HP);
      a1 = wmma16(af, bf, a1);
      v16h af2 = load_a_frag(Ao + kc, HP);
      v16h bf2 = load_b_frag(Whh2p + (t * 16) * HP + kc, HP);
      a2 = wmma16(af2, bf2, a2);
    }
#pragma unroll
    for (int v = 0; v < 8; ++v) {
      int m = v + hi * 8, n = t * 16 + nlo;
      Gi[m * G3P + n] = a1[v] + (n < G3 ? bih2[n] : 0.f);
      Gh[m * G3P + n] = a2[v] + (n < G3 ? bhh2[n] : 0.f);
    }
  }
  __syncthreads();
  for (int idx = lane; idx < 16 * HID; idx += 32) {
    int m = idx / HID, c = idx % HID;
    size_t g = (size_t)(row0 + m);
    float ir = Gi[m * G3P + c], iz = Gi[m * G3P + HID + c], in = Gi[m * G3P + 2 * HID + c];
    float hr = Gh[m * G3P + c], hz = Gh[m * G3P + HID + c], hn = Gh[m * G3P + 2 * HID + c];
    float r = sigm(ir + hr), z = sigm(iz + hz);
    float nn = tanhf(in + r * hn);
    float hp = outG[g * HID + c];
    outG[g * HID + c] = fmaxf((1.f - z) * nn + z * hp, 0.f);
  }
}

__global__ void final_kernel(const float* __restrict__ outG, const float* __restrict__ W2,
                             const float* __restrict__ b2, float* __restrict__ res){
  int g = blockIdx.x * blockDim.x + threadIdx.x;
  if (g >= NGRAPH) return;
  float s = b2[0];
  for (int c = 0; c < HID; ++c) s += outG[(size_t)g * HID + c] * W2[c];
  res[g] = s;
}

// ---------------- host launcher ----------------
extern "C" void kernel_launch(void* const* d_in, const int* in_sizes, int n_in,
                              void* d_out, int out_size, void* d_ws, size_t ws_size,
                              hipStream_t stream) {
  (void)in_sizes; (void)n_in; (void)out_size; (void)ws_size;
  const float* x         = (const float*)d_in[0];
  const int*   edge_index= (const int*)  d_in[1];
  const float* edge_attr = (const float*)d_in[2];
  const int*   batch     = (const int*)  d_in[3];
  const float* W1  = (const float*)d_in[4];
  const float* b1  = (const float*)d_in[5];
  const float* We1 = (const float*)d_in[6];
  const float* att_l = (const float*)d_in[7];
  const float* att_r = (const float*)d_in[8];
  const float* We2 = (const float*)d_in[9];
  const float* gate_bias = (const float*)d_in[10];
  const float* Wih1 = (const float*)d_in[11];
  const float* bih1 = (const float*)d_in[12];
  const float* Whh1 = (const float*)d_in[13];
  const float* bhh1 = (const float*)d_in[14];
  const float* Wm   = (const float*)d_in[15];
  const float* att_src = (const float*)d_in[16];
  const float* att_dst = (const float*)d_in[17];
  const float* mol_bias = (const float*)d_in[18];
  const float* Wih2 = (const float*)d_in[19];
  const float* bih2 = (const float*)d_in[20];
  const float* Whh2 = (const float*)d_in[21];
  const float* bhh2 = (const float*)d_in[22];
  const float* W2   = (const float*)d_in[23];
  const float* b2   = (const float*)d_in[24];

  char* base = (char*)d_ws;
  size_t off = 0;
  auto alloc = [&](size_t bytes) -> void* {
    off = (off + 255) & ~(size_t)255;
    void* p = base + off;
    off += bytes;
    return p;
  };
  h16* W1p   = (h16*)alloc((size_t)FINP * 64 * sizeof(h16));
  h16* We1p  = (h16*)alloc((size_t)HP * 64 * sizeof(h16));
  h16* We2p  = (h16*)alloc((size_t)HP * 64 * sizeof(h16));
  h16* Wih1p = (h16*)alloc((size_t)HP * G3P * sizeof(h16));
  h16* Whh1p = (h16*)alloc((size_t)HP * G3P * sizeof(h16));
  h16* Wmp   = (h16*)alloc((size_t)HP * 64 * sizeof(h16));
  h16* Wih2p = (h16*)alloc((size_t)HP * G3P * sizeof(h16));
  h16* Whh2p = (h16*)alloc((size_t)HP * G3P * sizeof(h16));
  float*    xh     = (float*)   alloc((size_t)NNODES * HID * 4);
  h16*      xh16   = (h16*)     alloc((size_t)NNODES * HP * 2);
  float*    ae     = (float*)   alloc((size_t)NEDGES * 4);
  unsigned* amaxN  = (unsigned*)alloc((size_t)NNODES * 4);
  float*    esumN  = (float*)   alloc((size_t)NNODES * 4);
  float*    hN     = (float*)   alloc((size_t)NNODES * HID * 4);
  float*    xl     = (float*)   alloc((size_t)NNODES * HID * 4);
  float*    al     = (float*)   alloc((size_t)NNODES * 4);
  float*    an     = (float*)   alloc((size_t)NNODES * 4);
  float*    outsum = (float*)   alloc((size_t)NGRAPH * HID * 4);
  float*    outG   = (float*)   alloc((size_t)NGRAPH * HID * 4);
  float*    arG    = (float*)   alloc((size_t)NGRAPH * 4);
  unsigned* amaxB  = (unsigned*)alloc((size_t)NGRAPH * 4);
  float*    esumB  = (float*)   alloc((size_t)NGRAPH * 4);
  float*    hmB    = (float*)   alloc((size_t)NGRAPH * HID * 4);

  auto cdiv = [](int a, int b){ return (a + b - 1) / b; };

  // weight packing (tiny)
  pack_weight_kernel<<<cdiv(FINP*64,256),256,0,stream>>>(W1,  W1p,  FIN, HID, FINP, 64);
  pack_weight_kernel<<<cdiv(HP*64,256),  256,0,stream>>>(We1, We1p, HID+EDIM, HID, HP, 64);
  pack_weight_kernel<<<cdiv(HP*64,256),  256,0,stream>>>(We2, We2p, HID, HID, HP, 64);
  pack_weight_kernel<<<cdiv(HP*G3P,256), 256,0,stream>>>(Wih1,Wih1p,HID, G3,  HP, G3P);
  pack_weight_kernel<<<cdiv(HP*G3P,256), 256,0,stream>>>(Whh1,Whh1p,HID, G3,  HP, G3P);
  pack_weight_kernel<<<cdiv(HP*64,256),  256,0,stream>>>(Wm,  Wmp,  HID, HID, HP, 64);
  pack_weight_kernel<<<cdiv(HP*G3P,256), 256,0,stream>>>(Wih2,Wih2p,HID, G3,  HP, G3P);
  pack_weight_kernel<<<cdiv(HP*G3P,256), 256,0,stream>>>(Whh2,Whh2p,HID, G3,  HP, G3P);

  init_node_kernel<<<cdiv(NNODES*HID,256),256,0,stream>>>(hN, gate_bias, amaxN, esumN, outsum);

  lin1_kernel<<<cdiv(NNODES,16),32,0,stream>>>(x, W1p, b1, xh, xh16);

  edge_a_kernel<<<cdiv(NEDGES,16),32,0,stream>>>(xh16, xh, edge_attr, edge_index,
                                                 We1p, att_l, att_r, ae, amaxN);
  edge_exp_kernel<<<cdiv(NEDGES,256),256,0,stream>>>(ae, edge_index, amaxN, esumN);
  edge_scatter_kernel<<<cdiv(NEDGES,16),32,0,stream>>>(xh16, edge_attr, edge_index,
                                                       We1p, We2p, ae, esumN, hN);

  gru1_kernel<<<cdiv(NNODES,16),32,0,stream>>>(hN, xh, xh16, Wih1p, bih1, Whh1p, bhh1,
                                               batch, outsum);
  relu_out_kernel<<<cdiv(NGRAPH*HID,256),256,0,stream>>>(outsum, outG);
  xl_kernel<<<cdiv(NNODES,16),32,0,stream>>>(xh16, Wmp, att_src, xl, al);

  for (int t = 0; t < TSTEPS; ++t) {
    tinit_kernel<<<cdiv(NGRAPH*HID,256),256,0,stream>>>(amaxB, esumB, hmB, mol_bias);
    ar_kernel<<<cdiv(NGRAPH,16),32,0,stream>>>(outG, Wmp, att_dst, arG);
    node_a_kernel<<<cdiv(NNODES,256),256,0,stream>>>(al, arG, batch, an, amaxB);
    node_exp_kernel<<<cdiv(NNODES,256),256,0,stream>>>(an, batch, amaxB, esumB);
    hm_kernel<<<cdiv(NNODES*HID,256),256,0,stream>>>(an, batch, esumB, xl, hmB);
    gru2_kernel<<<cdiv(NGRAPH,16),32,0,stream>>>(hmB, outG, Wih2p, bih2, Whh2p, bhh2);
  }

  final_kernel<<<cdiv(NGRAPH,256),256,0,stream>>>(outG, W2, b2, (float*)d_out);
}
